// PCENetwork_704374636922
// MI455X (gfx1250) — matrix operand order repaired and checked by reading, not compiled, not run on hardware
//
#include <hip/hip_runtime.h>
#include <math.h>

typedef __attribute__((ext_vector_type(16))) _Float16 v16h;
typedef __attribute__((ext_vector_type(8)))  float    v8f;

#define PS   8
#define EE   8
#define LL   2
#define CSH  8
#define NCLS 100
#define BB   64
#define HH   128
#define WW   128
#define PPI  256            // patches per image (16x16)
#define NTOK (BB*PPI)       // 16384 tokens
#define CAPT 2560           // ceil(1.25*N/E)
#define POSC 18
#define K72  72
#define PADH 832            // padded token stride in halves ([8ch][10][10] = 800, +32 slack)

__device__ __forceinline__ float silu_f(float x) { return x / (1.f + __expf(-x)); }
__device__ __forceinline__ float gelu_tanh(float x) {
    const float k0 = 0.7978845608028654f, k1 = 0.044715f;
    return 0.5f * x * (1.f + tanhf(k0 * (x + k1 * x * x * x)));
}

// ---------------------------------------------------------------------------
// 1) Stem: conv3x3 (3->8) + affine + SiLU, written directly in token layout
// ---------------------------------------------------------------------------
__global__ void stem_kernel(const float* __restrict__ X,
                            const float* __restrict__ W,
                            const float* __restrict__ scale,
                            const float* __restrict__ bias,
                            float* __restrict__ tok0)
{
    int idx = blockIdx.x * blockDim.x + threadIdx.x; // 64*8*128*128
    int x  = idx & 127;
    int y  = (idx >> 7) & 127;
    int co = (idx >> 14) & 7;
    int b  = idx >> 17;
    const float* Xb = X + (size_t)b * 3 * HH * WW;
    __builtin_prefetch(Xb + ((y + 4) & 127) * WW + x, 0, 1);
    float acc = 0.f;
#pragma unroll
    for (int ci = 0; ci < 3; ++ci) {
#pragma unroll
        for (int ky = 0; ky < 3; ++ky) {
            int yy = y + ky - 1;
            if ((unsigned)yy < HH) {
                const float* row = Xb + ci * HH * WW + yy * WW;
#pragma unroll
                for (int kx = 0; kx < 3; ++kx) {
                    int xx = x + kx - 1;
                    float v = ((unsigned)xx < WW) ? row[xx] : 0.f;
                    acc += v * W[(co * 3 + ci) * 9 + ky * 3 + kx];
                }
            }
        }
    }
    float r = silu_f(acc * scale[co] + bias[co]);
    int t = b * PPI + (y >> 3) * 16 + (x >> 3);
    tok0[((size_t)t * CSH + co) * 64 + (y & 7) * 8 + (x & 7)] = r;
}

// ---------------------------------------------------------------------------
// 2) Router (one block, thread p = patch p). meta per layer (1024 ints):
//    [0..255] e_idx, [256..511] rank, [512..519] counts, floats [768..1023] prob
// ---------------------------------------------------------------------------
__global__ void gate_kernel(const float* __restrict__ gate_w,
                            const float* __restrict__ gate_b,
                            int* __restrict__ meta,
                            float* __restrict__ out_scalars)
{
    __shared__ float psum[EE];
    __shared__ float zsum;
    __shared__ int   sh_e[PPI];
    int p  = threadIdx.x;
    int py = p >> 4, px = p & 15;

    float pp[POSC];
    {
        float sy = 0.f, sx = 0.f;
        float sY[4], cY[4], sX[4], cX[4];
#pragma unroll
        for (int k = 0; k < 4; ++k) { sY[k] = cY[k] = sX[k] = cX[k] = 0.f; }
        for (int i = 0; i < 8; ++i) {
            float ly = -1.f + (2.f / 127.f) * (float)(py * 8 + i);
            float lx = -1.f + (2.f / 127.f) * (float)(px * 8 + i);
            sy += ly; sx += lx;
#pragma unroll
            for (int k = 0; k < 4; ++k) {
                float fr = 3.14159265358979323846f * (float)(1 << k);
                sY[k] += sinf(fr * ly); cY[k] += cosf(fr * ly);
                sX[k] += sinf(fr * lx); cX[k] += cosf(fr * lx);
            }
        }
        pp[0] = sy * 0.125f; pp[1] = sx * 0.125f;
#pragma unroll
        for (int k = 0; k < 4; ++k) {
            pp[2 + 4 * k + 0] = sY[k] * 0.125f;
            pp[2 + 4 * k + 1] = cY[k] * 0.125f;
            pp[2 + 4 * k + 2] = sX[k] * 0.125f;
            pp[2 + 4 * k + 3] = cX[k] * 0.125f;
        }
    }

    float tz = 0.f, ti = 0.f;
    for (int l = 0; l < LL; ++l) {
        if (threadIdx.x == 0) { zsum = 0.f; for (int e = 0; e < EE; ++e) psum[e] = 0.f; }
        __syncthreads();

        float logit[EE]; float mx = -1e30f;
        for (int e = 0; e < EE; ++e) {
            float a = gate_b[l * EE + e];
            for (int c = 0; c < POSC; ++c) a += pp[c] * gate_w[(l * POSC + c) * EE + e];
            logit[e] = a; mx = fmaxf(mx, a);
        }
        float den = 0.f;
        for (int e = 0; e < EE; ++e) den += __expf(logit[e] - mx);
        float lse = mx + __logf(den);
        int be = 0; float bv = logit[0];
        for (int e = 1; e < EE; ++e) if (logit[e] > bv) { bv = logit[e]; be = e; }
        float pe = __expf(logit[be] - mx) / den;

        sh_e[p] = be;
        atomicAdd(&zsum, lse * lse);
        for (int e = 0; e < EE; ++e) atomicAdd(&psum[e], __expf(logit[e] - mx) / den);
        __syncthreads();

        int* me = meta + l * 1024;
        me[p] = be;
        ((float*)me)[768 + p] = pe;
        if (threadIdx.x == 0) {
            int cnt[EE]; for (int e = 0; e < EE; ++e) cnt[e] = 0;
            for (int q = 0; q < PPI; ++q) { int e = sh_e[q]; me[256 + q] = cnt[e]++; }
            for (int e = 0; e < EE; ++e) me[512 + e] = cnt[e];
            tz += zsum / (float)PPI;
            float imb = 0.f;
            for (int e = 0; e < EE; ++e) {
                int c = cnt[e]; int kept = 0;
                for (int b = 0; b < BB; ++b) {
                    int rem = CAPT - b * c;
                    int kb = rem < 0 ? 0 : (rem < c ? rem : c);
                    kept += kb;
                }
                imb += (psum[e] / (float)PPI) * ((float)kept / (float)NTOK);
            }
            ti += (float)EE * imb;
        }
        __syncthreads();
    }
    if (threadIdx.x == 0) { out_scalars[0] = tz; out_scalars[1] = ti; }
}

// ---------------------------------------------------------------------------
// 3) MoE layer: one wave32 per token.
//    - token copied global->LDS via global_load_async_to_lds_b128 (ASYNCcnt)
//    - re-laid as zero-halo padded f16 [c][10][10]: branch-free im2col taps
//    - expert weights staged per block as f16, zero-padded [e][16][96]:
//      B fragments are 16 contiguous halves (no masks)
//    - expert conv = WMMA f32_16x16x32_f16, M=64,K=72->96,N=8->16
//    - LAYER 0: fused per-token GroupNorm; LAYER 1: per-(img,ch) sum/sumsq
// ---------------------------------------------------------------------------
template<int LAYER>
__global__ void __launch_bounds__(256)
layer_kernel(const float* __restrict__ tokIn,
             float* __restrict__ tokOut,
             const float* __restrict__ exp_w,
             const float* __restrict__ exp_b,
             const float* __restrict__ gn_s,
             const float* __restrict__ gn_b,
             const int* __restrict__ meta,
             float* __restrict__ accS,
             float* __restrict__ accQ)
{
    __shared__ float    raw [8 * 512];        // 16 KB raw f32 tokens (async dst)
    __shared__ _Float16 pad [8 * PADH];       // 13 KB padded f16 tokens
    __shared__ _Float16 wlds[EE * 16 * 96];   // 24 KB f16 weights, zero-padded

    int tid = threadIdx.x;
    int ln  = tid & 31;
    int w   = tid >> 5;
    int t   = blockIdx.x * 8 + w;
    int p   = t & (PPI - 1);
    int b   = t >> 8;
    const int* me = meta + LAYER * 1024;

    // ---- kick off async token copy: 4 x 512B, each lane moves 16B ----
    {
        const float* src = tokIn + (size_t)t * 512;
        unsigned lbase = (unsigned)(size_t)(&raw[w * 512]);
#pragma unroll
        for (int i = 0; i < 4; ++i) {
            const float* g = src + i * 128 + ln * 4;
            unsigned la = lbase + (unsigned)(i * 512 + ln * 16);
            asm volatile("global_load_async_to_lds_b128 %0, %1, off"
                         :: "v"(la), "v"(g) : "memory");
        }
    }

    // ---- cooperative weight staging: zero, then fill ----
    {
        unsigned* wz = (unsigned*)wlds;
        for (int i = tid; i < EE * 16 * 96 / 2; i += 256) wz[i] = 0u;
    }
    __syncthreads();
    {
        const float* wsrc = exp_w + (size_t)LAYER * EE * CSH * K72;
        for (int i = tid; i < EE * CSH * K72; i += 256) {
            int e  = i / (CSH * K72);
            int r  = i - e * (CSH * K72);
            int co = r / K72;
            int K  = r - co * K72;
            wlds[(e * 16 + co) * 96 + K] = (_Float16)wsrc[i];
        }
    }

    // ---- wait for async copy, build padded-halo f16 token ----
    asm volatile("s_wait_asynccnt 0x0" ::: "memory");
    _Float16* tp = &pad[w * PADH];
    {
        unsigned* pz = (unsigned*)tp;
#pragma unroll
        for (int i = 0; i < 13; ++i) pz[ln * 13 + i] = 0u;   // zero 832 halves
#pragma unroll
        for (int i = 0; i < 16; ++i) {
            int v = ln * 16 + i;
            float f = raw[w * 512 + v];
            int c = v >> 6, r = (v >> 3) & 7, x = v & 7;
            tp[c * 100 + (r + 1) * 10 + (x + 1)] = (_Float16)f;
        }
    }
    __syncthreads();   // weights visible block-wide

    // ---- routing ----
    int e   = me[p];
    int rk  = me[256 + p];
    int c_e = me[512 + e];
    float pr = ((const float*)me)[768 + p];
    int slot = b * c_e + rk;
    float g = (slot < CAPT) ? pr : 0.f;

    int hi  = (ln >> 4) & 1;
    int col = ln & 15;

    // ---- B fragments: 16 contiguous halves each, zero-padded ----
    v16h bf[3];
#pragma unroll
    for (int kc = 0; kc < 3; ++kc) {
        const _Float16* wb = &wlds[(e * 16 + col) * 96 + kc * 32 + hi * 16];
        v16h bv;
#pragma unroll
        for (int j = 0; j < 16; ++j) bv[j] = wb[j];
        bf[kc] = bv;
    }

    // ---- A build (constant offsets into padded halo) + WMMA ----
    v8f vz = {};
    v8f cacc[4] = {vz, vz, vz, vz};
#pragma unroll
    for (int mt = 0; mt < 4; ++mt) {
        int m    = mt * 16 + col;               // A row = pixel
        int abase = (m >> 3) * 10 + (m & 7);    // halo coords, taps add +ty*10+tx
#pragma unroll
        for (int kc = 0; kc < 3; ++kc) {
            v16h a;
#pragma unroll
            for (int j = 0; j < 16; ++j) {
                const int K0 = kc * 32 + j;  // hi==0 K
                const int K1 = K0 + 16;                            // hi==1 K
                const int o0 = (K0 < K72)
                    ? ((K0 / 9) * 100 + ((K0 % 9) / 3) * 10 + (K0 % 9) % 3) : 0;
                const int o1 = (K1 < K72)
                    ? ((K1 / 9) * 100 + ((K1 % 9) / 3) * 10 + (K1 % 9) % 3) : 0;
                const bool c0 = (K0 < K72), c1 = (K1 < K72);
                int off = hi ? o1 : o0;
                _Float16 v;
                if (c0 && c1) {
                    v = tp[abase + off];
                } else if (!c0 && !c1) {
                    v = (_Float16)0.f;
                } else {  // mixed per lane-half -> single cndmask, no branch
                    _Float16 ld = tp[abase + off];
                    v = (hi ? c1 : c0) ? ld : (_Float16)0.f;
                }
                a[j] = v;
            }
            cacc[mt] = __builtin_amdgcn_wmma_f32_16x16x32_f16(
                false, a, false, bf[kc], (short)0, cacc[mt], false, false);
        }
    }

    // ---- epilogue ----
    float bias = (col < CSH) ? exp_b[(LAYER * EE + e) * CSH + col] : 0.f;
    float vals[4][8];
    float s1 = 0.f, s2 = 0.f;
#pragma unroll
    for (int mt = 0; mt < 4; ++mt)
#pragma unroll
        for (int rr = 0; rr < 8; ++rr) {
            float v = silu_f(cacc[mt][rr] + bias) * g;
            s1 += v; s2 += v * v;
            if (LAYER == 0) vals[mt][rr] = v;
        }

    if (LAYER == 0) {
        // per-token GroupNorm: group {2g,2g+1} lives in lanes {2g,2g+1,+16}
        s1 += __shfl_xor(s1, 1, 32);  s2 += __shfl_xor(s2, 1, 32);
        s1 += __shfl_xor(s1, 16, 32); s2 += __shfl_xor(s2, 16, 32);
        float mean = s1 * (1.f / 128.f);
        float var  = s2 * (1.f / 128.f) - mean * mean;
        float inv  = __frsqrt_rn(var + 1e-5f);
        if (col < CSH) {
            float sc = gn_s[LAYER * CSH + col], sb = gn_b[LAYER * CSH + col];
#pragma unroll
            for (int mt = 0; mt < 4; ++mt)
#pragma unroll
                for (int rr = 0; rr < 8; ++rr) {
                    int m = mt * 16 + rr + hi * 8;  // C/D: M = r + 8*hi
                    tokOut[((size_t)t * CSH + col) * 64 + m] =
                        (vals[mt][rr] - mean) * inv * sc + sb;
                }
        }
    } else {
        s1 += __shfl_xor(s1, 16, 32);
        s2 += __shfl_xor(s2, 16, 32);
        if (ln < CSH)                       unsafeAtomicAdd(&accS[b * CSH + ln], s1);
        else if (ln >= 16 && ln < 16 + CSH) unsafeAtomicAdd(&accQ[b * CSH + (ln - 16)], s2);
    }
}

// ---------------------------------------------------------------------------
// 4) Head: GN(layer1)+pool from channel sums, then MLP 8->32 (gelu) ->100
// ---------------------------------------------------------------------------
__global__ void head_kernel(const float* __restrict__ accS,
                            const float* __restrict__ accQ,
                            const float* __restrict__ gn_s,
                            const float* __restrict__ gn_b,
                            const float* __restrict__ w1, const float* __restrict__ b1,
                            const float* __restrict__ w2, const float* __restrict__ b2,
                            float* __restrict__ out)
{
    __shared__ float pooled[BB * CSH];
    __shared__ float h1[BB * 4 * CSH];
    int tid = threadIdx.x;
    const float NP = 16384.f; // 128*128 pixels per channel per image
    for (int i = tid; i < BB * CSH; i += blockDim.x) {
        int bb = i >> 3, c = i & 7, gch = c >> 1;
        float sa = accS[bb * CSH + 2 * gch], sb2 = accS[bb * CSH + 2 * gch + 1];
        float qa = accQ[bb * CSH + 2 * gch], qb  = accQ[bb * CSH + 2 * gch + 1];
        float meanC = accS[i] / NP;
        float mu  = (sa + sb2) / (2.f * NP);
        float var = (qa + qb) / (2.f * NP) - mu * mu;
        float inv = __frsqrt_rn(var + 1e-5f);
        pooled[i] = gn_s[CSH + c] * (meanC - mu) * inv + gn_b[CSH + c];
    }
    __syncthreads();
    for (int i = tid; i < BB * 4 * CSH; i += blockDim.x) {
        int bb = i >> 5, j = i & 31;
        float a = b1[j];
        for (int c = 0; c < CSH; ++c) a += pooled[bb * CSH + c] * w1[c * 32 + j];
        h1[i] = gelu_tanh(a);
    }
    __syncthreads();
    for (int i = tid; i < BB * NCLS; i += blockDim.x) {
        int bb = i / NCLS, k = i - bb * NCLS;
        float a = b2[k];
        for (int j = 0; j < 32; ++j) a += h1[bb * 32 + j] * w2[j * NCLS + k];
        out[i] = a;
    }
}

// ---------------------------------------------------------------------------
extern "C" void kernel_launch(void* const* d_in, const int* in_sizes, int n_in,
                              void* d_out, int out_size, void* d_ws, size_t ws_size,
                              hipStream_t stream)
{
    const float* X      = (const float*)d_in[0];
    const float* stem_w = (const float*)d_in[1];
    const float* stem_s = (const float*)d_in[2];
    const float* stem_b = (const float*)d_in[3];
    const float* gate_w = (const float*)d_in[4];
    const float* gate_b = (const float*)d_in[5];
    const float* exp_w  = (const float*)d_in[6];
    const float* exp_b  = (const float*)d_in[7];
    const float* gn_s   = (const float*)d_in[8];
    const float* gn_b   = (const float*)d_in[9];
    const float* hw1    = (const float*)d_in[10];
    const float* hb1    = (const float*)d_in[11];
    const float* hw2    = (const float*)d_in[12];
    const float* hb2    = (const float*)d_in[13];
    float* out = (float*)d_out;

    char* ws = (char*)d_ws;
    const size_t tokBytes = (size_t)NTOK * 512 * sizeof(float); // 33.5 MB
    float* tok0 = (float*)ws;
    float* tok1 = (float*)(ws + tokBytes);
    int*   meta = (int*)  (ws + 2 * tokBytes);
    float* accS = (float*)(ws + 2 * tokBytes + 8192);
    float* accQ = accS + BB * CSH;

    hipMemsetAsync(accS, 0, 2 * BB * CSH * sizeof(float), stream);

    stem_kernel<<<(BB * CSH * HH * WW) / 256, 256, 0, stream>>>(
        X, stem_w, stem_s, stem_b, tok0);
    gate_kernel<<<1, 256, 0, stream>>>(gate_w, gate_b, meta, out + BB * NCLS);
    layer_kernel<0><<<NTOK / 8, 256, 0, stream>>>(
        tok0, tok1, exp_w, exp_b, gn_s, gn_b, meta, nullptr, nullptr);
    layer_kernel<1><<<NTOK / 8, 256, 0, stream>>>(
        tok1, nullptr, exp_w, exp_b, gn_s, gn_b, meta, accS, accQ);
    head_kernel<<<1, 256, 0, stream>>>(
        accS, accQ, gn_s, gn_b, hw1, hb1, hw2, hb2, out);
}
